// FullKP_41549513622108
// MI455X (gfx1250) — matrix operand order repaired and checked by reading, not compiled, not run on hardware
//
#include <hip/hip_runtime.h>
#include <hip/hip_bf16.h>

typedef __attribute__((ext_vector_type(16))) _Float16 v16h;
typedef __attribute__((ext_vector_type(8)))  _Float16 v8h;
typedef __attribute__((ext_vector_type(8)))  float    v8f;

#define KCH  100   // last-layer output channels (5*5*4)
#define CMAX 128   // padded cout

// ---------------------------------------------------------------------------
// Repack f32 weights (Cout,Cin,3,3) -> f16 [9 taps][128 co][CinPad ci], zero pad
// ---------------------------------------------------------------------------
__global__ void pack_weights_k(const float* __restrict__ w, _Float16* __restrict__ dst,
                               int Cout, int Cin, int CinPad) {
  int idx = blockIdx.x * blockDim.x + threadIdx.x;
  int total = 9 * CMAX * CinPad;
  if (idx >= total) return;
  int t  = idx / (CMAX * CinPad);
  int r  = idx - t * (CMAX * CinPad);
  int co = r / CinPad;
  int ci = r - co * CinPad;
  float v = 0.f;
  if (co < Cout && ci < Cin) {
    int dy = t / 3, dx = t % 3;
    v = w[((co * Cin + ci) * 3 + dy) * 3 + dx];
  }
  dst[idx] = (_Float16)v;
}

// ---------------------------------------------------------------------------
// Build layer-0 input: [2][144][144][64] f16 (ch 0..2 = gf, 3..34 = feats, pad 0)
// ---------------------------------------------------------------------------
__global__ void pack_input_k(const float* __restrict__ feat,   // [2][4][8][144][144]
                             const float* __restrict__ gf,     // [2][3][1][1]
                             _Float16* __restrict__ dst) {
  int idx = blockIdx.x * blockDim.x + threadIdx.x;
  int total = 2 * 144 * 144 * 64;
  if (idx >= total) return;
  int c = idx & 63;
  int p = idx >> 6;
  int x = p % 144;  p /= 144;
  int y = p % 144;
  int b = p / 144;
  float v = 0.f;
  if (c < 3) {
    v = gf[b * 3 + c];
  } else if (c < 35) {
    int j = c - 3;                       // j = s*8 + f
    v = feat[(((size_t)(b * 4 + (j >> 3)) * 8 + (j & 7)) * 144 + y) * 144 + x];
  }
  dst[idx] = (_Float16)v;
}

// ---------------------------------------------------------------------------
// 3x3 VALID conv as implicit GEMM on WMMA f16->f32, templated on CinPad so all
// fragment offsets fold into load-immediate fields.
//   block = 256 threads (8 waves); tile = 64 pixels (one output row) x 128 cout
//   wave w -> cout tile [16w,16w+16), iterates 4 M-tiles sharing one B fragment
// ---------------------------------------------------------------------------
template <int CINPAD>
__global__ __launch_bounds__(256)
void conv3x3_wmma_k(const _Float16* __restrict__ actIn,  // [bs][Hin][Win][CINPAD]
                    const _Float16* __restrict__ wts,    // [9][128][CINPAD] f16
                    const float* __restrict__ bias,      // [Cout] f32
                    _Float16* __restrict__ actOut,       // [bs][Hout][Wout][128]
                    float* __restrict__ logitsOut,       // [bs][Hout][Wout][KCH]
                    int Hin, int Win, int Cout, int last) {
  const int Hout = Hin - 2, Wout = Win - 2;
  const int b  = blockIdx.z;
  const int y  = blockIdx.y;
  const int x0 = blockIdx.x * 64;

  __shared__ _Float16 lds[3 * 66 * CINPAD];

  // ---- async-stage 3 rows x 66 cols x CINPAD into LDS (covers all 9 taps) ----
  {
    const int CH = CINPAD >> 3;               // 16B chunks per pixel
    const int totalChunks = 3 * 66 * CH;
    for (int t = threadIdx.x; t < totalChunks; t += 256) {
      int row = t / (66 * CH);
      int rem = t - row * (66 * CH);
      int col = rem / CH;
      int cc  = rem - col * CH;
      int gx = x0 + col; if (gx > Win - 1) gx = Win - 1;   // clamp right edge
      int gy = y + row;
      const _Float16* gp = actIn + (size_t)((b * Hin + gy) * Win + gx) * CINPAD + cc * 8;
      unsigned ldsOff = (unsigned)(uintptr_t)(const void*)&lds[t * 8];
      unsigned long long gaddr = (unsigned long long)(uintptr_t)gp;
      asm volatile("global_load_async_to_lds_b128 %0, %1, off"
                   :: "v"(ldsOff), "v"(gaddr) : "memory");
    }
    asm volatile("s_wait_asynccnt 0x0" ::: "memory");
  }
  __syncthreads();

  const int wave = threadIdx.x >> 5;
  const int lane = threadIdx.x & 31;
  const int lm   = lane & 15;
  const int hi   = lane >> 4;           // lane half selects K sub-ranges
  const int nb   = wave * 16;           // this wave's cout tile

  // Per-lane fragment bases; all tap/kstep/mtile offsets are immediates.
  // A(m,k):  lds[(dy*66 + m0+lm+dx)*CINPAD + kb + hi*8 (+16)]
  // B(k,n):  wts[(tap*128 + nb+lm)*CINPAD + kb + hi*8 (+16)]
  const _Float16* aBase = lds + (size_t)lm * CINPAD + hi * 8;
  const _Float16* bBase = wts + (size_t)(nb + lm) * CINPAD + hi * 8;

  v8f acc[4] = {};

  #pragma unroll
  for (int dy = 0; dy < 3; ++dy) {
    #pragma unroll
    for (int dx = 0; dx < 3; ++dx) {
      #pragma unroll
      for (int ks = 0; ks < CINPAD / 32; ++ks) {
        const int kb = ks * 32;
        // one B fragment for this (tap, kstep), reused by 4 M-tiles
        const _Float16* bp = bBase + (size_t)((dy * 3 + dx) * CMAX) * CINPAD + kb;
        v16h bf;
        *(v8h*)&bf         = *(const v8h*)bp;
        *(((v8h*)&bf) + 1) = *(const v8h*)(bp + 16);
        // batch all 4 A fragments so the ds_loads issue as one clause
        v16h afr[4];
        #pragma unroll
        for (int mt = 0; mt < 4; ++mt) {
          const _Float16* ap = aBase + (size_t)(dy * 66 + dx + mt * 16) * CINPAD + kb;
          *(v8h*)&afr[mt]         = *(const v8h*)ap;
          *(((v8h*)&afr[mt]) + 1) = *(const v8h*)(ap + 16);
        }
        #pragma unroll
        for (int mt = 0; mt < 4; ++mt) {
          acc[mt] = __builtin_amdgcn_wmma_f32_16x16x32_f16(
              false, afr[mt], false, bf, (short)0, acc[mt], false, false);
        }
      }
    }
  }

  // ---- epilogue: bias (+ leaky ReLU -> f16) or f32 logits ----
  int co = nb + lm;
  bool cok = co < Cout;
  float bv = cok ? bias[co] : 0.f;
  #pragma unroll
  for (int mt = 0; mt < 4; ++mt) {
    #pragma unroll
    for (int r = 0; r < 8; ++r) {
      int x = x0 + mt * 16 + r + hi * 8;  // C tile: vgpr r -> row r / r+8 by lane half
      if (cok && x < Wout) {
        float v = acc[mt][r] + bv;
        if (!last) {
          v = v > 0.f ? v : 0.01f * v;
          actOut[(size_t)((b * Hout + y) * Wout + x) * CMAX + co] = (_Float16)v;
        } else {
          logitsOut[(size_t)((b * Hout + y) * Wout + x) * KCH + co] = v;
        }
      }
    }
  }
}

// ---------------------------------------------------------------------------
// Per-pixel: softmax over 100 logits, apply 5x5 kernels per spp to radiance.
//   logits : [2][126][126][100], rad : [2][4][3][144][144], out : [2][3][126][126]
// ---------------------------------------------------------------------------
__global__ void softmax_apply_k(const float* __restrict__ logits,
                                const float* __restrict__ rad,
                                float* __restrict__ out) {
  const int HW = 126 * 126;
  int idx = blockIdx.x * blockDim.x + threadIdx.x;
  if (idx >= 2 * HW) return;
  int b = idx / HW;
  int p = idx - b * HW;
  int y = p / 126, x = p - (p / 126) * 126;

  const float* lg = logits + (size_t)idx * KCH;
  float mx = lg[0];
  for (int i = 1; i < KCH; ++i) mx = fmaxf(mx, lg[i]);
  float sum = 0.f;
  for (int i = 0; i < KCH; ++i) sum += __expf(lg[i] - mx);
  float inv = 1.f / sum;

  float o0 = 0.f, o1 = 0.f, o2 = 0.f;
  for (int s = 0; s < 4; ++s) {
    const float* rb = rad + (size_t)(b * 4 + s) * 3 * 144 * 144;
    #pragma unroll
    for (int t = 0; t < 25; ++t) {
      int ti = t / 5, tj = t - (t / 5) * 5;
      float kv = __expf(lg[s * 25 + t] - mx) * inv;
      int ro = (y + 7 + ti) * 144 + (x + 7 + tj);
      o0 += kv * rb[ro];
      o1 += kv * rb[144 * 144 + ro];
      o2 += kv * rb[2 * 144 * 144 + ro];
    }
  }
  out[(size_t)(b * 3 + 0) * HW + p] = o0;
  out[(size_t)(b * 3 + 1) * HW + p] = o1;
  out[(size_t)(b * 3 + 2) * HW + p] = o2;
}

// ---------------------------------------------------------------------------
extern "C" void kernel_launch(void* const* d_in, const int* in_sizes, int n_in,
                              void* d_out, int out_size, void* d_ws, size_t ws_size,
                              hipStream_t stream) {
  (void)in_sizes; (void)n_in; (void)out_size; (void)ws_size;

  const float* features = (const float*)d_in[0];
  const float* radiance = (const float*)d_in[1];
  const float* gfeat    = (const float*)d_in[2];
  const float* W[9]; const float* B[9];
  for (int i = 0; i < 9; ++i) { W[i] = (const float*)d_in[3 + i]; B[i] = (const float*)d_in[12 + i]; }

  // ---- bump allocate workspace ----
  char* ws = (char*)d_ws;
  size_t off = 0;
  auto balloc = [&](size_t bytes) -> void* {
    void* pp = ws + off;
    off += (bytes + 255) & ~(size_t)255;
    return pp;
  };

  int cinPad[9]; cinPad[0] = 64; for (int i = 1; i < 9; ++i) cinPad[i] = 128;
  int cin[9];    cin[0]    = 35; for (int i = 1; i < 9; ++i) cin[i]    = 128;
  int cout[9];   for (int i = 0; i < 8; ++i) cout[i] = 128; cout[8] = KCH;

  _Float16* wbuf[9];
  for (int i = 0; i < 9; ++i)
    wbuf[i] = (_Float16*)balloc((size_t)9 * CMAX * cinPad[i] * sizeof(_Float16));
  _Float16* actA  = (_Float16*)balloc((size_t)2 * 144 * 144 * CMAX * sizeof(_Float16));
  _Float16* actB  = (_Float16*)balloc((size_t)2 * 144 * 144 * CMAX * sizeof(_Float16));
  float*    logit = (float*)   balloc((size_t)2 * 126 * 126 * KCH  * sizeof(float));

  // ---- repack weights to f16 tap-major ----
  for (int i = 0; i < 9; ++i) {
    int total = 9 * CMAX * cinPad[i];
    pack_weights_k<<<(total + 255) / 256, 256, 0, stream>>>(W[i], wbuf[i],
                                                            cout[i], cin[i], cinPad[i]);
  }
  // ---- build layer-0 input ----
  {
    int total = 2 * 144 * 144 * 64;
    pack_input_k<<<(total + 255) / 256, 256, 0, stream>>>(features, gfeat, actA);
  }
  // ---- conv chain (ping-pong) ----
  for (int l = 0; l < 9; ++l) {
    int Hin = 144 - 2 * l, Win = Hin;
    int Hout = Hin - 2, Wout = Win - 2;
    const _Float16* inp = (l & 1) ? actB : actA;
    _Float16* outp      = (l & 1) ? actA : actB;
    dim3 grid((Wout + 63) / 64, Hout, 2);
    if (l == 0) {
      conv3x3_wmma_k<64><<<grid, 256, 0, stream>>>(inp, wbuf[l], B[l], outp, logit,
                                                   Hin, Win, cout[l], 0);
    } else {
      conv3x3_wmma_k<128><<<grid, 256, 0, stream>>>(inp, wbuf[l], B[l], outp, logit,
                                                    Hin, Win, cout[l], (l == 8) ? 1 : 0);
    }
  }
  // ---- softmax + kernel apply ----
  {
    int total = 2 * 126 * 126;
    softmax_apply_k<<<(total + 255) / 256, 256, 0, stream>>>(logit, radiance, (float*)d_out);
  }
}
